// NeuralODE_53472342835317
// MI455X (gfx1250) — matrix operand order repaired and checked by reading, not compiled, not run on hardware
//
#include <hip/hip_runtime.h>
#include <hip/hip_bf16.h>
#include <math.h>

typedef __attribute__((ext_vector_type(16))) _Float16 v16h;
typedef __attribute__((ext_vector_type(8)))  float    v8f;

#define HDIM 128
#define NIN  32
#define NOUT 31
#define ITEMS 8
#define NTOT (256*256)

__device__ __forceinline__ v8f wmma_f16(v16h a, v16h b, v8f c) {
  // D = A(16x32 f16) * B(32x16 f16) + C(16x16 f32)
  return __builtin_amdgcn_wmma_f32_16x16x32_f16(false, a, false, b, (short)0, c, false, false);
}

// A fragment: 16x32 tile from row-major [rows][128] f16 matrix in LDS.
// Per ISA 7.12.2: lanes 0-15 row M=lane, K in {0..7,16..23}; lanes 16-31 K in {8..15,24..31}.
__device__ __forceinline__ v16h ldsA(const _Float16* base, int row0, int k0) {
  int lane = threadIdx.x & 31;
  const _Float16* p = base + (row0 + (lane & 15)) * HDIM + k0;
  int hi = (lane >> 4) << 3;                 // 0 or 8
  v16h a;
  uint32_t* au = (uint32_t*)&a;
  const uint32_t* pu = (const uint32_t*)p;   // k0 multiple of 32 -> dword aligned
#pragma unroll
  for (int q = 0; q < 8; ++q) {
    int K = ((q & 4) << 2) + hi + ((q & 3) << 1);  // q<4: hi+2q ; q>=4: 16+hi+2(q-4)
    au[q] = pu[K >> 1];
  }
  return a;
}

// B fragment: 32x16 tile; B kept TRANSPOSED in LDS as [n][k] (ld=128) so the
// K-halves per lane are contiguous. Lanes 0-15: K in {0..7,16..23}; 16-31: K in {8..15,24..31}.
__device__ __forceinline__ v16h ldsB(const _Float16* baseT, int n0, int k0) {
  int lane = threadIdx.x & 31;
  const _Float16* p = baseT + (n0 + (lane & 15)) * HDIM + k0 + ((lane >> 4) << 3);
  v16h b;
  uint32_t* bu = (uint32_t*)&b;
  const uint32_t* pu = (const uint32_t*)p;
#pragma unroll
  for (int q = 0; q < 8; ++q) bu[q] = pu[(q < 4) ? q : q + 4];
  return b;
}

__global__ __launch_bounds__(256)
void node_fused(const float* __restrict__ tptr, const float* __restrict__ y,
                const float* __restrict__ W0, const float* __restrict__ b0,
                const float* __restrict__ W1, const float* __restrict__ b1,
                const float* __restrict__ W2, const float* __restrict__ b2,
                const float* __restrict__ W3, const float* __restrict__ b3,
                float* __restrict__ out, float* __restrict__ jac) {
  __shared__ _Float16 lW1h[HDIM*HDIM];   // 32 KB, row-major [m][k]
  __shared__ _Float16 lW2h[HDIM*HDIM];   // 32 KB
  __shared__ _Float16 lW0T[NIN*HDIM];    // 8 KB, [n][k] = W0[k][n] (full, incl force col)
  __shared__ _Float16 lW3A[NIN*HDIM];    // 8 KB, [m][k], row 31 zero-padded
  __shared__ _Float16 lMa[NIN*HDIM];     // 8 KB, B operand transposed [n][k]
  __shared__ _Float16 lMb[NIN*HDIM];     // 8 KB
  __shared__ float    lH[3][ITEMS*HDIM]; // 12 KB activations
  __shared__ float    lX[ITEMS*NIN];     // 1 KB

  const int t    = threadIdx.x;
  const int i0   = blockIdx.x * ITEMS;
  const int wave = t >> 5;
  const int lane = t & 31;

  // ---- stage weights as f16 ----
  for (int i = t; i < HDIM*HDIM; i += 256) {
    lW1h[i] = (_Float16)W1[i];
    lW2h[i] = (_Float16)W2[i];
  }
  for (int i = t; i < NIN*HDIM; i += 256) {
    int n = i >> 7, k = i & (HDIM-1);
    lW0T[i] = (_Float16)W0[k*NIN + n];
    lW3A[i] = (n < NOUT) ? (_Float16)W3[n*HDIM + k] : (_Float16)0.f;
  }
  // ---- inputs: x = [y, sin(t)] ----
  {
    int item = t >> 5, n = t & 31;
    float force = sinf(tptr[0]);
    lX[t] = (n < NOUT) ? y[(size_t)(i0 + item)*NOUT + n] : force;
  }
  __syncthreads();

  // ---- forward pass (VALU; ~3% of FLOPs). Skewed k to avoid bank conflicts. ----
  for (int u = 0; u < 4; ++u) {
    int idx = t + 256*u; int item = idx >> 7; int m = idx & (HDIM-1);
    float acc = b0[m];
    const float* x = &lX[item*NIN];
#pragma unroll
    for (int n = 0; n < NIN; ++n) acc += (float)lW0T[n*HDIM + m] * x[n];
    lH[0][idx] = tanhf(acc);
  }
  __syncthreads();
  for (int u = 0; u < 4; ++u) {
    int idx = t + 256*u; int item = idx >> 7; int m = idx & (HDIM-1);
    float acc = b1[m];
    const float* hin = &lH[0][item*HDIM];
    for (int k = 0; k < HDIM; ++k) { int kk = (k + m) & (HDIM-1); acc += (float)lW1h[m*HDIM + kk] * hin[kk]; }
    lH[1][idx] = tanhf(acc);
  }
  __syncthreads();
  for (int u = 0; u < 4; ++u) {
    int idx = t + 256*u; int item = idx >> 7; int m = idx & (HDIM-1);
    float acc = b2[m];
    const float* hin = &lH[1][item*HDIM];
    for (int k = 0; k < HDIM; ++k) { int kk = (k + m) & (HDIM-1); acc += (float)lW2h[m*HDIM + kk] * hin[kk]; }
    lH[2][idx] = tanhf(acc);
  }
  __syncthreads();
  if (t < ITEMS*NOUT) {
    int item = t / NOUT, m = t % NOUT;
    float acc = b3[m];
    const float* hin = &lH[2][item*HDIM];
    for (int k = 0; k < HDIM; ++k) { int kk = (k + m) & (HDIM-1); acc += (float)lW3A[m*HDIM + kk] * hin[kk]; }
    out[(size_t)(i0 + item)*NOUT + m] = acc;
  }

  // ---- Jacobian: J = W3 * D2 * W2 * D1 * W1 * D0 * W0[:, :31], one item at a time ----
  for (int it = 0; it < ITEMS; ++it) {
    __syncthreads();
    // M0^T[n][k] = (1-h0[k]^2) * W0[k][n] for n<31, col 31 (force) dropped -> 0
    for (int i = t; i < NIN*HDIM; i += 256) {
      int n = i >> 7, k = i & (HDIM-1);
      float h = lH[0][it*HDIM + k];
      float v = (n < NOUT) ? (1.f - h*h) * (float)lW0T[i] : 0.f;
      lMa[i] = (_Float16)v;
    }
    __syncthreads();

    // GEMM1: M1 = D1 * (W1 @ M0) -> lMb ; wave w owns M-tile w, both N-tiles (A reuse)
    {
      v8f c0 = {}; v8f c1 = {};
#pragma unroll
      for (int ks = 0; ks < 4; ++ks) {
        v16h a   = ldsA(lW1h, wave*16, ks*32);
        v16h bb0 = ldsB(lMa, 0,  ks*32);
        v16h bb1 = ldsB(lMa, 16, ks*32);
        c0 = wmma_f16(a, bb0, c0);
        c1 = wmma_f16(a, bb1, c1);
      }
      int m0 = wave*16 + ((lane >> 4) << 3);
      int n  = lane & 15;
      alignas(16) _Float16 t0[8]; alignas(16) _Float16 t1[8];
#pragma unroll
      for (int r = 0; r < 8; ++r) {
        float h = lH[1][it*HDIM + m0 + r];
        float d = 1.f - h*h;
        t0[r] = (_Float16)(c0[r]*d);
        t1[r] = (_Float16)(c1[r]*d);
      }
      uint32_t* dst = (uint32_t*)&lMb[n*HDIM + m0];
      const uint32_t* s = (const uint32_t*)t0;
#pragma unroll
      for (int q = 0; q < 4; ++q) dst[q] = s[q];
      dst = (uint32_t*)&lMb[(16 + n)*HDIM + m0];
      s = (const uint32_t*)t1;
#pragma unroll
      for (int q = 0; q < 4; ++q) dst[q] = s[q];
    }
    __syncthreads();

    // GEMM2: M2 = D2 * (W2 @ M1) -> lMa
    {
      v8f c0 = {}; v8f c1 = {};
#pragma unroll
      for (int ks = 0; ks < 4; ++ks) {
        v16h a   = ldsA(lW2h, wave*16, ks*32);
        v16h bb0 = ldsB(lMb, 0,  ks*32);
        v16h bb1 = ldsB(lMb, 16, ks*32);
        c0 = wmma_f16(a, bb0, c0);
        c1 = wmma_f16(a, bb1, c1);
      }
      int m0 = wave*16 + ((lane >> 4) << 3);
      int n  = lane & 15;
      alignas(16) _Float16 t0[8]; alignas(16) _Float16 t1[8];
#pragma unroll
      for (int r = 0; r < 8; ++r) {
        float h = lH[2][it*HDIM + m0 + r];
        float d = 1.f - h*h;
        t0[r] = (_Float16)(c0[r]*d);
        t1[r] = (_Float16)(c1[r]*d);
      }
      uint32_t* dst = (uint32_t*)&lMa[n*HDIM + m0];
      const uint32_t* s = (const uint32_t*)t0;
#pragma unroll
      for (int q = 0; q < 4; ++q) dst[q] = s[q];
      dst = (uint32_t*)&lMa[(16 + n)*HDIM + m0];
      s = (const uint32_t*)t1;
#pragma unroll
      for (int q = 0; q < 4; ++q) dst[q] = s[q];
    }
    __syncthreads();

    // GEMM3: J = W3 @ M2 -> global (waves 0..3: 2x2 tiles of 32x32)
    if (wave < 4) {
      int mt = wave >> 1, nt = wave & 1;
      v8f c = {};
#pragma unroll
      for (int ks = 0; ks < 4; ++ks) {
        v16h a  = ldsA(lW3A, mt*16, ks*32);
        v16h bb = ldsB(lMa, nt*16, ks*32);
        c = wmma_f16(a, bb, c);
      }
      int m0 = mt*16 + ((lane >> 4) << 3);
      int n  = nt*16 + (lane & 15);
      if (n < NOUT) {
        float* jrow = jac + (size_t)(i0 + it)*(NOUT*NOUT) + n;
#pragma unroll
        for (int r = 0; r < 8; ++r) {
          int m = m0 + r;
          if (m < NOUT) jrow[(size_t)m*NOUT] = c[r];
        }
      }
    }
  }
}

extern "C" void kernel_launch(void* const* d_in, const int* in_sizes, int n_in,
                              void* d_out, int out_size, void* d_ws, size_t ws_size,
                              hipStream_t stream) {
  const float* tp = (const float*)d_in[0];
  const float* y  = (const float*)d_in[1];
  const float* W0 = (const float*)d_in[2];
  const float* b0 = (const float*)d_in[3];
  const float* W1 = (const float*)d_in[4];
  const float* b1 = (const float*)d_in[5];
  const float* W2 = (const float*)d_in[6];
  const float* b2 = (const float*)d_in[7];
  const float* W3 = (const float*)d_in[8];
  const float* b3 = (const float*)d_in[9];
  float* out = (float*)d_out;
  float* jac = out + (size_t)NTOT*NOUT;   // concatenated tuple output: [out | jac]
  dim3 grid(NTOT / ITEMS);
  node_fused<<<grid, 256, 0, stream>>>(tp, y, W0, b0, W1, b1, W2, b2, W3, b3, out, jac);
}
